// AmericanOptionRNN_18769007083815
// MI455X (gfx1250) — compile-verified, matrix operand order checked by reading, not run on hardware
//
#include <hip/hip_runtime.h>

#define B_ 8192
#define T_ 128
#define D_ 16
#define H_ 32
#define EPS_ 1e-5f
#define NWAVES 8   // 256-thread block = 8 wave32, each owns a 16-row batch tile

typedef _Float16 h16;
typedef __attribute__((ext_vector_type(16))) _Float16 v16h;
typedef __attribute__((ext_vector_type(8)))  _Float16 v8h;
typedef __attribute__((ext_vector_type(8)))  float    v8f;

struct NetParams {
  const float *Wih0,*Whh0,*bih0,*bhh0,*Wih1,*Whh1,*bih1,*bhh1;
  const float *embW,*embB,*bng,*bnb,*bnm,*bnv,*outW,*outB;
};
struct AllParams { NetParams net[2]; };

// Fast sigmoid: v_exp + v_rcp (1-ulp class), fine vs the fp16-GEMM noise floor.
__device__ __forceinline__ float sigmf(float x) {
  return __builtin_amdgcn_rcpf(1.0f + __expf(-x));
}
// tanh(x) = 2*sigmoid(2x) - 1  (avoids the branchy OCML tanh polynomial)
__device__ __forceinline__ float tanh_fast(float x) {
  return __builtin_fmaf(2.0f, sigmf(2.0f * x), -1.0f);
}
__device__ __forceinline__ v8f splat8(float x) {
  v8f v = { x, x, x, x, x, x, x, x };
  return v;
}

// Re-stripe a 16x32 matrix held in C/D layout (per lane: 8 rows x 2 cols) into the
// A-matrix 16x32 f16 fragment layout, via a per-wave LDS bounce buffer.
// A layout: lane l -> row l%16; halves 0..7 = K base 8*(l>>4), halves 8..15 = 16 + 8*(l>>4).
// Only one fence needed: LDS ops are in-order within a wave, so the next call's
// stores cannot pass this call's loads in the DS pipe (and aliasing blocks compiler reorder).
__device__ __forceinline__ v16h d2a(const v8f& c0, const v8f& c1, h16* buf, int lane) {
  const int hi = lane >> 4, lo = lane & 15;
#pragma unroll
  for (int vr = 0; vr < 8; ++vr) {
    const int row = hi * 8 + vr;
    buf[row * 40 + lo]      = (h16)c0[vr];
    buf[row * 40 + 16 + lo] = (h16)c1[vr];
  }
  asm volatile("s_wait_dscnt 0" ::: "memory");   // cross-lane store->load visibility
  const int arow = lane & 15;
  const int kb   = (lane >> 4) * 8;
  v8h l8 = *(const v8h*)(buf + arow * 40 + kb);
  v8h h8 = *(const v8h*)(buf + arow * 40 + 16 + kb);
  return __builtin_shufflevector(l8, h8, 0,1,2,3,4,5,6,7,8,9,10,11,12,13,14,15);
}

// One GRU cell for a 16-row tile. xA/hA are A-fragments; weights are prepacked
// B-fragments in LDS (6 tiles of 16 cols covering the 96 gate outputs r|z|n).
// Gate biases ride in on the WMMA C operand (lane-splat, constant over the 8 rows).
// State is carried in C/D layout (hc0: cols lo, hc1: cols 16+lo).
__device__ __forceinline__ void gru_step(int lane, const v16h& xA, const v16h& hA,
                                         const h16* wI, const h16* wH,
                                         const float* bi, const float* bh,
                                         v8f& hc0, v8f& hc1) {
  v8f gi[6], gh[6];
#pragma unroll
  for (int n = 0; n < 6; ++n) {
    v16h wi = *(const v16h*)(wI + (n * 32 + lane) * 16);
    v16h wh = *(const v16h*)(wH + (n * 32 + lane) * 16);
    gi[n] = __builtin_amdgcn_wmma_f32_16x16x32_f16(false, xA, false, wi, (short)0, splat8(bi[n]), false, false);
    gh[n] = __builtin_amdgcn_wmma_f32_16x16x32_f16(false, hA, false, wh, (short)0, splat8(bh[n]), false, false);
  }
#pragma unroll
  for (int c = 0; c < 2; ++c) {
    v8f& hc = c ? hc1 : hc0;
#pragma unroll
    for (int j = 0; j < 8; ++j) {
      float r  = sigmf(gi[c][j] + gh[c][j]);                       // r = sig(ir+hr)  (biases in C)
      float z  = sigmf(gi[2 + c][j] + gh[2 + c][j]);               // z = sig(iz+hz)
      float nn = tanh_fast(__builtin_fmaf(r, gh[4 + c][j], gi[4 + c][j])); // n = tanh(inn + r*hn)
      hc[j] = __builtin_fmaf(1.0f - z, nn, z * hc[j]);             // (1-z)*n + z*h
    }
  }
}

__global__ __launch_bounds__(256)
void american_rnn_kernel(const float* __restrict__ S, const float* __restrict__ gS,
                         const float* __restrict__ hp0, const float* __restrict__ hd0,
                         AllParams prm, float* __restrict__ out) {
  // Weight B-fragments: 27 frags x 32 lanes x 16 halves (~27 KB).
  __shared__ __align__(32) h16 wlds[27 * 32 * 16];
  __shared__ __align__(32) h16 tbuf[NWAVES][16 * 40];

  const int tid  = threadIdx.x;
  const int lane = tid & 31;
  const int wid  = tid >> 5;
  const int net  = blockIdx.y;           // 0 = price net, 1 = delta net
  const int lo   = lane & 15, hi = lane >> 4;
  const NetParams P = prm.net[net];
  const int outRows = net ? D_ : 1;

  // ---- Prepack weight B-fragments (fp32 -> fp16) once per block (wave 0) ----
  // B 32x16 layout mirrors A: lane -> col N=lo; halves = W^T[kb0.. | kb1.., N]
  // = W[N_row, k] contiguous in k.  Frag ids: 0-5 Wih0, 6-11 Whh0, 12-17 Wih1,
  // 18-23 Whh1, 24-25 embW, 26 outW (rows >= outRows zero-padded).
  if (wid == 0) {
    const float* Ws[5] = { P.Wih0, P.Whh0, P.Wih1, P.Whh1, P.embW };
    const int    nt[5] = { 6, 6, 6, 6, 2 };
    const int kb0 = hi * 8, kb1 = 16 + hi * 8;
    int fid = 0;
    for (int m = 0; m < 5; ++m) {
      for (int t0 = 0; t0 < nt[m]; ++t0, ++fid) {
        const float* src = Ws[m] + (t0 * 16 + lo) * 32;
        h16* dst = wlds + (fid * 32 + lane) * 16;
#pragma unroll
        for (int j = 0; j < 8; ++j) { dst[j] = (h16)src[kb0 + j]; dst[8 + j] = (h16)src[kb1 + j]; }
      }
    }
    { // outW: (outRows x 32), zero-pad cols >= outRows
      const bool ok = lo < outRows;
      const float* src = P.outW + lo * 32;
      h16* dst = wlds + (26 * 32 + lane) * 16;
#pragma unroll
      for (int j = 0; j < 8; ++j) {
        dst[j]     = ok ? (h16)src[kb0 + j] : (h16)0.0f;
        dst[8 + j] = ok ? (h16)src[kb1 + j] : (h16)0.0f;
      }
    }
  }
  __syncthreads();

  // ---- Per-lane scalar constants (gate biases, folded BN affine) ----
  float bi0[6], bh0[6], bi1[6], bh1[6];
#pragma unroll
  for (int n = 0; n < 6; ++n) {
    bi0[n] = P.bih0[16 * n + lo]; bh0[n] = P.bhh0[16 * n + lo];
    bi1[n] = P.bih1[16 * n + lo]; bh1[n] = P.bhh1[16 * n + lo];
  }
  float alpha[2], beta[2];
#pragma unroll
  for (int c = 0; c < 2; ++c) {
    const int col = 16 * c + lo;
    const float rs = rsqrtf(P.bnv[col] + EPS_) * P.bng[col];
    alpha[c] = rs;
    beta[c]  = (P.embB[col] - P.bnm[col]) * rs + P.bnb[col];   // fold emb bias + BN shift
  }
  const float outBl = (lo < outRows) ? P.outB[lo] : 0.0f;

  // ---- Initial hidden state in C/D layout ----
  const float* h0src = net ? hd0 : hp0;      // shape [2, B, H]
  const int b0 = (blockIdx.x * NWAVES + wid) * 16;
  v8f h0c0, h0c1, h1c0, h1c1;
#pragma unroll
  for (int vr = 0; vr < 8; ++vr) {
    const size_t b = (size_t)(b0 + hi * 8 + vr);
    h0c0[vr] = h0src[b * H_ + lo];
    h0c1[vr] = h0src[b * H_ + 16 + lo];
    h1c0[vr] = h0src[((size_t)B_ + b) * H_ + lo];
    h1c1[vr] = h0src[((size_t)B_ + b) * H_ + 16 + lo];
  }

  h16* tb = tbuf[wid];
  v16h h0A = d2a(h0c0, h0c1, tb, lane);
  v16h h1A = d2a(h1c0, h1c1, tb, lane);

  float* prices = out;                       // [B, T, 1]
  float* deltas = out + (size_t)B_ * T_;     // [B, T, D]

  const int arow = lane & 15;
  const int kb   = (lane >> 4) * 8;

  // Backward time loop (t = T-1 .. 0); output index == t after the flip-back.
  for (int t = T_ - 1; t >= 0; --t) {
    // x_t A-fragment: K<16 -> S row slice (contiguous), K>=16 -> broadcast g_S.
    const float* sp = S + ((size_t)(b0 + arow) * T_ + t) * D_ + kb;
    const float gsv = gS[(size_t)(b0 + arow) * T_ + t];
    v16h xA;
#pragma unroll
    for (int j = 0; j < 8; ++j) { xA[j] = (h16)sp[j]; xA[8 + j] = (h16)gsv; }

    // Layer 0: gi = x Wih0^T + bih0, gh = h0 Whh0^T + bhh0
    gru_step(lane, xA, h0A, wlds, wlds + 6 * 512, bi0, bh0, h0c0, h0c1);
    h0A = d2a(h0c0, h0c1, tb, lane);
    // Layer 1: gi = h0 Wih1^T + bih1, gh = h1 Whh1^T + bhh1
    gru_step(lane, h0A, h1A, wlds + 12 * 512, wlds + 18 * 512, bi1, bh1, h1c0, h1c1);
    h1A = d2a(h1c0, h1c1, tb, lane);

    // Head (fused): e = h1 embW^T -> BN affine -> SiLU -> o = e outW^T + outB
    const v8f zero = splat8(0.0f);
    v16h we0 = *(const v16h*)(wlds + (24 * 32 + lane) * 16);
    v16h we1 = *(const v16h*)(wlds + (25 * 32 + lane) * 16);
    v8f e0 = __builtin_amdgcn_wmma_f32_16x16x32_f16(false, h1A, false, we0, (short)0, zero, false, false);
    v8f e1 = __builtin_amdgcn_wmma_f32_16x16x32_f16(false, h1A, false, we1, (short)0, zero, false, false);
#pragma unroll
    for (int j = 0; j < 8; ++j) {
      float v0 = __builtin_fmaf(e0[j], alpha[0], beta[0]); e0[j] = v0 * sigmf(v0);  // SiLU
      float v1 = __builtin_fmaf(e1[j], alpha[1], beta[1]); e1[j] = v1 * sigmf(v1);
    }
    v16h eA = d2a(e0, e1, tb, lane);
    v16h wo = *(const v16h*)(wlds + (26 * 32 + lane) * 16);
    v8f o = __builtin_amdgcn_wmma_f32_16x16x32_f16(false, eA, false, wo, (short)0, splat8(outBl), false, false);

    if (net == 0) {
      if (lo == 0) {
#pragma unroll
        for (int vr = 0; vr < 8; ++vr)
          prices[(size_t)(b0 + hi * 8 + vr) * T_ + t] = o[vr];
      }
    } else {
#pragma unroll
      for (int vr = 0; vr < 8; ++vr)
        deltas[((size_t)(b0 + hi * 8 + vr) * T_ + t) * D_ + lo] = sigmf(o[vr]);
    }
  }
}

extern "C" void kernel_launch(void* const* d_in, const int* in_sizes, int n_in,
                              void* d_out, int out_size, void* d_ws, size_t ws_size,
                              hipStream_t stream) {
  (void)in_sizes; (void)n_in; (void)out_size; (void)d_ws; (void)ws_size;
  const float* S   = (const float*)d_in[0];
  const float* gS  = (const float*)d_in[1];
  const float* hp0 = (const float*)d_in[5];
  const float* hd0 = (const float*)d_in[6];
  AllParams prm;
  for (int n = 0; n < 2; ++n) {
    const float** p = (const float**)&prm.net[n];
    const int base = 7 + n * 16;   // p_* at 7..22, d_* at 23..38 (setup_inputs order)
    for (int k = 0; k < 16; ++k) p[k] = (const float*)d_in[base + k];
  }
  dim3 grid(B_ / (16 * NWAVES), 2, 1);   // (64, 2): x = batch tiles, y = net select
  american_rnn_kernel<<<grid, 256, 0, stream>>>(S, gS, hp0, hd0, prm, (float*)d_out);
}
